// CpGPredictor_63711544869462
// MI455X (gfx1250) — matrix operand order repaired and checked by reading, not compiled, not run on hardware
//
#include <hip/hip_runtime.h>

// ---------------------------------------------------------------------------
// Fused 2-layer LSTM (H=64, gate order i,f,g,o) + final FC, persistent kernel.
// Workgroup = 128 threads (4 wave32) handles 16 batch rows for all T steps.
// Wave w owns hidden units [16w,16w+16); computes its 4 gate tiles via
// v_wmma_f32_16x16x32_bf16 (bf16 A/B, f32 accum). h exchanged via ping-pong
// LDS; c kept in registers in WMMA C/D layout; weights register-resident bf16.
// x is staged time-major in LDS, software-pipelined through registers, with
// global_prefetch_b8 warming L2 two chunks ahead.
// ---------------------------------------------------------------------------

#define HDIM 64

typedef __attribute__((ext_vector_type(16))) __bf16 v16bf;
typedef __attribute__((ext_vector_type(8)))  __bf16 v8bf;
typedef __attribute__((ext_vector_type(8)))  float  v8f;

static __device__ __forceinline__ float sigmoid_fast(float x) {
  return __builtin_amdgcn_rcpf(1.0f + __expf(-x));
}
static __device__ __forceinline__ float tanh_fast(float x) {
  return 2.0f * __builtin_amdgcn_rcpf(1.0f + __expf(-2.0f * x)) - 1.0f;
}

static __device__ __forceinline__ v8f wmma_bf16(v16bf a, v16bf b, v8f c) {
  // (neg_a, A, neg_b, B, c_mod, C, reuse_a, reuse_b)
  return __builtin_amdgcn_wmma_f32_16x16x32_bf16(false, a, false, b, (short)0, c,
                                                 false, false);
}

// B operand (32x16) for gates = h @ W^T:  B[k][n] = W[nrow][k].
// Lane n (0..15 / 16..31=p), element j -> K = p*8 + j (j<8), 16 + p*8 + (j-8) (j>=8).
static __device__ __forceinline__ v16bf load_w_tile(const float* __restrict__ W,
                                                    int nrow, int kbase) {
  const float* s = W + nrow * HDIM + kbase;
  v16bf b;
#pragma unroll
  for (int j = 0; j < 16; ++j) b[j] = (__bf16)s[j < 8 ? j : j + 8];
  return b;
}

// A operand (16x32) from row-major LDS h[16][64] bf16.
// Lane row = lane&15, half p = lane>>4: a[0..7] = h[row][kc*32 + p*8 + j],
// a[8..15] = h[row][kc*32 + 16 + p*8 + j]  -> two aligned 16B LDS loads.
static __device__ __forceinline__ v16bf load_a_tile(const __bf16* hbuf, int row,
                                                    int kc, int p) {
  const __bf16* s = hbuf + row * HDIM + kc * 32 + p * 8;
  v8bf lo = *(const v8bf*)s;
  v8bf hi = *(const v8bf*)(s + 16);
  return __builtin_shufflevector(lo, hi, 0, 1, 2, 3, 4, 5, 6, 7, 8, 9, 10, 11,
                                 12, 13, 14, 15);
}

__global__ __launch_bounds__(128) void lstm2_fused_kernel(
    const float* __restrict__ x,     // [B,T]
    const float* __restrict__ Wih0,  // [256,1]
    const float* __restrict__ Whh0,  // [256,64]
    const float* __restrict__ bih0, const float* __restrict__ bhh0,
    const float* __restrict__ Wih1,  // [256,64]
    const float* __restrict__ Whh1,  // [256,64]
    const float* __restrict__ bih1, const float* __restrict__ bhh1,
    const float* __restrict__ fcW,   // [1,64]
    const float* __restrict__ fcb,   // [1]
    float* __restrict__ out,         // [B]
    int T) {
  __shared__ __align__(16) __bf16 sH1[2][16 * HDIM];  // ping-pong layer-0 h
  __shared__ __align__(16) __bf16 sH2[2][16 * HDIM];  // ping-pong layer-1 h
  __shared__ __align__(16) float sX[32][16];          // x chunk, TIME-major
  __shared__ float sHF[16 * HDIM];                    // final h2 (f32) for FC

  const int tid  = threadIdx.x;
  const int lane = tid & 31;
  const int w    = tid >> 5;    // wave 0..3 -> owns units [16w,16w+16)
  const int n    = lane & 15;   // column within the wave's 16 units (== A row)
  const int p    = lane >> 4;   // lane half
  const int u    = w * 16 + n;  // global hidden unit for this lane's column
  const int b0   = blockIdx.x * 16;

  // Zero ping-pong h buffers (h(-1) == 0).
  {
    __bf16* z1 = &sH1[0][0];
    __bf16* z2 = &sH2[0][0];
    for (int i = tid; i < 2 * 16 * HDIM; i += 128) {
      z1[i] = (__bf16)0.0f;
      z2[i] = (__bf16)0.0f;
    }
  }

  // Per-lane scalar constants: combined biases and the rank-1 input weights.
  const float bi0 = bih0[u] + bhh0[u];
  const float bf0 = bih0[u + 64] + bhh0[u + 64];
  const float bg0 = bih0[u + 128] + bhh0[u + 128];
  const float bo0 = bih0[u + 192] + bhh0[u + 192];
  const float wi0 = Wih0[u], wf0 = Wih0[u + 64], wg0 = Wih0[u + 128],
              wo0 = Wih0[u + 192];
  const float bi1 = bih1[u] + bhh1[u];
  const float bf1 = bih1[u + 64] + bhh1[u + 64];
  const float bg1 = bih1[u + 128] + bhh1[u + 128];
  const float bo1 = bih1[u + 192] + bhh1[u + 192];

  // Register-resident bf16 weight tiles in native B-operand layout.
  // [gate][k-chunk]; nrow = 64*gate + 16*w + n; kbase = 32*kc + 8*p.
  v16bf Bhh0[4][2], Bih1[4][2], Bhh1[4][2];
#pragma unroll
  for (int g = 0; g < 4; ++g) {
#pragma unroll
    for (int kc = 0; kc < 2; ++kc) {
      const int nrow = g * 64 + w * 16 + n;
      const int kbase = kc * 32 + p * 8;
      Bhh0[g][kc] = load_w_tile(Whh0, nrow, kbase);
      Bih1[g][kc] = load_w_tile(Wih1, nrow, kbase);
      Bhh1[g][kc] = load_w_tile(Whh1, nrow, kbase);
    }
  }

  v8f c1 = {};      // layer-0 cell state, element v -> (row v+8p, unit u)
  v8f c2 = {};      // layer-1 cell state
  v8f h2last = {};  // last h2 in f32 for the FC head

  // Software pipeline for x: thread i covers (row=i>>5, t-offset=i&31).
  const int xr_row = tid >> 5;        // rows handled: xr_row, +4, +8, +12
  const int xr_tt  = tid & 31;
  float xr[4];
#pragma unroll
  for (int q = 0; q < 4; ++q)
    xr[q] = x[(size_t)(b0 + xr_row + 4 * q) * T + xr_tt];

  __syncthreads();

#pragma unroll 1
  for (int tc = 0; tc < T; tc += 32) {
    // Commit the register-staged chunk to LDS (transposed to time-major).
#pragma unroll
    for (int q = 0; q < 4; ++q) sX[xr_tt][xr_row + 4 * q] = xr[q];
    // Issue loads for the next chunk; latency hidden behind 32 steps.
    if (tc + 32 < T) {
#pragma unroll
      for (int q = 0; q < 4; ++q)
        xr[q] = x[(size_t)(b0 + xr_row + 4 * q) * T + (tc + 32) + xr_tt];
    }
    // Warm L2 two chunks ahead (one 128B line == one chunk per batch row).
    if (tid < 16 && tc + 64 < T)
      __builtin_prefetch(&x[(size_t)(b0 + tid) * T + tc + 64], 0, 0);
    __syncthreads();

#pragma unroll 1
    for (int ts = 0; ts < 32; ++ts) {
      const int t = tc + ts;
      const int pp = t & 1;
      const __bf16* h1p = sH1[pp];      // h1[t-1]
      const __bf16* h2p = sH2[pp];      // h2[t-1]
      __bf16* h1c = sH1[pp ^ 1];        // h1[t]
      __bf16* h2c = sH2[pp ^ 1];        // h2[t]

      // ---------------- layer 0 ----------------
      const v16bf a00 = load_a_tile(h1p, n, 0, p);
      const v16bf a01 = load_a_tile(h1p, n, 1, p);

      // Rows 8p..8p+7 of this timestep's x: two contiguous 16B LDS loads.
      const float4 xlo = *(const float4*)&sX[ts][8 * p];
      const float4 xhi = *(const float4*)&sX[ts][8 * p + 4];
      const float xv[8] = {xlo.x, xlo.y, xlo.z, xlo.w,
                           xhi.x, xhi.y, xhi.z, xhi.w};

      v8f ai, af, ag, ao;
#pragma unroll
      for (int v = 0; v < 8; ++v) {
        ai[v] = xv[v] * wi0 + bi0;
        af[v] = xv[v] * wf0 + bf0;
        ag[v] = xv[v] * wg0 + bg0;
        ao[v] = xv[v] * wo0 + bo0;
      }
      ai = wmma_bf16(a00, Bhh0[0][0], ai); ai = wmma_bf16(a01, Bhh0[0][1], ai);
      af = wmma_bf16(a00, Bhh0[1][0], af); af = wmma_bf16(a01, Bhh0[1][1], af);
      ag = wmma_bf16(a00, Bhh0[2][0], ag); ag = wmma_bf16(a01, Bhh0[2][1], ag);
      ao = wmma_bf16(a00, Bhh0[3][0], ao); ao = wmma_bf16(a01, Bhh0[3][1], ao);

      v8f h1new;
#pragma unroll
      for (int v = 0; v < 8; ++v) {
        const float ii = sigmoid_fast(ai[v]);
        const float ff = sigmoid_fast(af[v]);
        const float gg = tanh_fast(ag[v]);
        const float oo = sigmoid_fast(ao[v]);
        const float c = ff * c1[v] + ii * gg;
        c1[v] = c;
        h1new[v] = oo * tanh_fast(c);
      }
#pragma unroll
      for (int v = 0; v < 8; ++v)
        h1c[(v + 8 * p) * HDIM + u] = (__bf16)h1new[v];
      __syncthreads();  // h1[t] visible to all waves

      // ---------------- layer 1 ----------------
      const v16bf a10 = load_a_tile(h1c, n, 0, p);
      const v16bf a11 = load_a_tile(h1c, n, 1, p);
      const v16bf a20 = load_a_tile(h2p, n, 0, p);
      const v16bf a21 = load_a_tile(h2p, n, 1, p);

      v8f gi, gf, gg_, go;
#pragma unroll
      for (int v = 0; v < 8; ++v) {
        gi[v] = bi1; gf[v] = bf1; gg_[v] = bg1; go[v] = bo1;
      }
      gi = wmma_bf16(a10, Bih1[0][0], gi); gi = wmma_bf16(a11, Bih1[0][1], gi);
      gi = wmma_bf16(a20, Bhh1[0][0], gi); gi = wmma_bf16(a21, Bhh1[0][1], gi);
      gf = wmma_bf16(a10, Bih1[1][0], gf); gf = wmma_bf16(a11, Bih1[1][1], gf);
      gf = wmma_bf16(a20, Bhh1[1][0], gf); gf = wmma_bf16(a21, Bhh1[1][1], gf);
      gg_ = wmma_bf16(a10, Bih1[2][0], gg_); gg_ = wmma_bf16(a11, Bih1[2][1], gg_);
      gg_ = wmma_bf16(a20, Bhh1[2][0], gg_); gg_ = wmma_bf16(a21, Bhh1[2][1], gg_);
      go = wmma_bf16(a10, Bih1[3][0], go); go = wmma_bf16(a11, Bih1[3][1], go);
      go = wmma_bf16(a20, Bhh1[3][0], go); go = wmma_bf16(a21, Bhh1[3][1], go);

#pragma unroll
      for (int v = 0; v < 8; ++v) {
        const float ii = sigmoid_fast(gi[v]);
        const float ff = sigmoid_fast(gf[v]);
        const float gg = tanh_fast(gg_[v]);
        const float oo = sigmoid_fast(go[v]);
        const float c = ff * c2[v] + ii * gg;
        c2[v] = c;
        const float h = oo * tanh_fast(c);
        h2last[v] = h;
        h2c[(v + 8 * p) * HDIM + u] = (__bf16)h;
      }
      __syncthreads();  // h2[t] (and h1 ping-pong reuse) safe for next step
    }
  }

  // ---------------- FC head on h2[:, T-1, :] ----------------
#pragma unroll
  for (int v = 0; v < 8; ++v) sHF[(v + 8 * p) * HDIM + u] = h2last[v];
  __syncthreads();

  if (tid < 16) {
    float s = fcb[0];
#pragma unroll 4
    for (int k = 0; k < HDIM; ++k) s += sHF[tid * HDIM + k] * fcW[k];
    out[b0 + tid] = s;
  }
}

extern "C" void kernel_launch(void* const* d_in, const int* in_sizes, int n_in,
                              void* d_out, int out_size, void* d_ws,
                              size_t ws_size, hipStream_t stream) {
  const float* x    = (const float*)d_in[0];
  const float* Wih0 = (const float*)d_in[1];
  const float* Whh0 = (const float*)d_in[2];
  const float* bih0 = (const float*)d_in[3];
  const float* bhh0 = (const float*)d_in[4];
  const float* Wih1 = (const float*)d_in[5];
  const float* Whh1 = (const float*)d_in[6];
  const float* bih1 = (const float*)d_in[7];
  const float* bhh1 = (const float*)d_in[8];
  const float* fcW  = (const float*)d_in[9];
  const float* fcb  = (const float*)d_in[10];
  float* out = (float*)d_out;

  const int B = out_size;             // 1024
  const int T = in_sizes[0] / B;      // 2048 (multiple of 32)

  dim3 grid(B / 16), block(128);
  hipLaunchKernelGGL(lstm2_fused_kernel, grid, block, 0, stream, x, Wih0, Whh0,
                     bih0, bhh0, Wih1, Whh1, bih1, bhh1, fcW, fcb, out, T);
}
